// SlotAttention_24223615549899
// MI455X (gfx1250) — matrix operand order repaired and checked by reading, not compiled, not run on hardware
//
#include <hip/hip_runtime.h>
#include <hip/hip_bf16.h>

typedef __attribute__((ext_vector_type(16))) _Float16 v16h;
typedef __attribute__((ext_vector_type(8)))  float    v8f;

namespace {
constexpr int kB = 32, kS = 16, kN = 4096, kD = 256, kM = 1024;
}

// ---------------------------------------------------------------------------
// WMMA helper (CDNA5 v_wmma_f32_16x16x32_f16)
// ---------------------------------------------------------------------------
__device__ __forceinline__ v8f wmma16(v16h a, v16h b, v8f c) {
  // (neg_a, A, neg_b, B, c_mod, C, reuse_a, reuse_b)
  return __builtin_amdgcn_wmma_f32_16x16x32_f16(false, a, false, b, (short)0, c,
                                                false, false);
}

// ---------------------------------------------------------------------------
// Fragment-packed storage. A tile (16x32) or B tile (32x16) of f16 occupies
// 512 halves laid out as (lane 0..31) x (16 halves), so a fragment load is a
// single contiguous 32B per-lane vector load. These index helpers are the
// (m|n, k) -> offset maps for a contiguous k order, k = 16*(lane/16) + i.
// ---------------------------------------------------------------------------
// A 16x32 (MxK): lanes 0-15: M=lane, K=0..15; lanes16-31: M=lane-16,
// K=16..31.
__device__ __forceinline__ int a_frag_off(int m, int kk) {  // kk in 0..31
  const int lane = m + (kk & 16);
  const int i = kk & 15;
  return lane * 16 + i;
}
// B 32x16 (KxN): lanes 0-15: N=lane, K=0..15; lanes 16-31: N=lane-16, K=16..31.
__device__ __forceinline__ int b_frag_off(int kk, int c) {  // kk 0..31, c 0..15
  return ((kk & 16) + c) * 16 + (kk & 15);
}

__device__ __forceinline__ v16h load_frag(const _Float16* p, int tile, int lane) {
  return *(const v16h*)(p + ((size_t)tile << 9) + (lane << 4));
}

// B = K^T fragment for logits: B(k,n) = Kmat[n0+n][k0+k]; per-lane contiguous.
__device__ __forceinline__ v16h load_bt_f16(const _Float16* kmat, int ld, int n0,
                                            int k0, int lane) {
  const int n = n0 + (lane & 15);
  const int koff = (lane & 16);
  return *(const v16h*)(kmat + (size_t)n * ld + k0 + koff);
}

// A fragment built from f32 row-major source with on-the-fly f16 convert
// (used only for the small GRU GEMMs whose A comes from f32 atomics).
__device__ __forceinline__ v16h load_a_f32(const float* base, int ld, int lane) {
  const int m = lane & 15;
  const int kb = (lane & 16) ? 16 : 0;
  const float* row = base + m * ld;
  v16h a;
#pragma unroll
  for (int v = 0; v < 8; ++v) {
    const int k = 2 * v + kb;
    a[2 * v]     = (_Float16)row[k];
    a[2 * v + 1] = (_Float16)row[k + 1];
  }
  return a;
}

// ---------------------------------------------------------------------------
// Block-cooperative LayerNorm of 16 rows x kD, written to LDS directly in
// A-fragment-packed order (8 ktiles x 512 halves). 128 threads.
// ---------------------------------------------------------------------------
__device__ void layernorm16_to_lds_pack(const float* src, const float* gamma,
                                        const float* beta, _Float16* dst,
                                        float* red) {
  const int t = threadIdx.x;  // 8 threads per row
  const int row = t >> 3;
  const int sub = t & 7;
  const int chunk = kD / 8;   // 32
  const float* p = src + row * kD + sub * chunk;
  float s = 0.f, ss = 0.f;
#pragma unroll 8
  for (int i = 0; i < chunk; ++i) { float x = p[i]; s += x; ss += x * x; }
  red[(row * 8 + sub) * 2 + 0] = s;
  red[(row * 8 + sub) * 2 + 1] = ss;
  __syncthreads();
  float mean = 0.f, m2 = 0.f;
#pragma unroll
  for (int j = 0; j < 8; ++j) {
    mean += red[(row * 8 + j) * 2 + 0];
    m2   += red[(row * 8 + j) * 2 + 1];
  }
  mean *= (1.f / kD);
  const float var = m2 * (1.f / kD) - mean * mean;
  const float rstd = rsqrtf(var + 1e-6f);
#pragma unroll 8
  for (int i = 0; i < chunk; ++i) {
    const int d = sub * chunk + i;
    const float y = (p[i] - mean) * rstd * gamma[d] + beta[d];
    dst[((d >> 5) << 9) + a_frag_off(row, d & 31)] = (_Float16)y;
  }
  __syncthreads();
}

// ---------------------------------------------------------------------------
// Kernel 0: repack a [K x N] f32 weight into f16 B-fragment-tiled order.
// ---------------------------------------------------------------------------
__global__ void pack_b_kernel(const float* __restrict__ src,
                              _Float16* __restrict__ dst, int K, int N) {
  int idx = blockIdx.x * blockDim.x + threadIdx.x;
  if (idx >= K * N) return;
  const int k = idx / N, n = idx % N;
  const int tile = (k >> 5) * (N >> 4) + (n >> 4);
  dst[((size_t)tile << 9) + b_frag_off(k & 31, n & 15)] = (_Float16)src[idx];
}

// ---------------------------------------------------------------------------
// Kernel 1: LN(inputs) fused with K/V projections.  One block = 16 rows.
// K stored row-major f16 (consumed as contiguous K^T fragments).
// V stored in B-fragment order (consumed as B tiles of the updates GEMM).
// ---------------------------------------------------------------------------
__global__ __launch_bounds__(128) void ln_kv_kernel(
    const float* __restrict__ inputs, const float* __restrict__ g,
    const float* __restrict__ be, const _Float16* __restrict__ wkp,
    const _Float16* __restrict__ wvp, _Float16* __restrict__ kf,
    _Float16* __restrict__ vpack) {
  __shared__ _Float16 At[8 * 512];
  __shared__ float red[16 * 8 * 2];
  const int rb = blockIdx.x;  // 0 .. B*N/16
  const int b = rb / (kN / 16);
  const int n0 = (rb % (kN / 16)) * 16;
  layernorm16_to_lds_pack(inputs + ((size_t)b * kN + n0) * kD, g, be, At, red);
  const int wave = threadIdx.x >> 5, lane = threadIdx.x & 31;
#pragma unroll
  for (int t = 0; t < 4; ++t) {
    const int nt = wave * 4 + t;  // 16 d-tiles over 4 waves
    v8f ck = {};
    v8f cv = {};
#pragma unroll
    for (int ks = 0; ks < 8; ++ks) {
      v16h a  = load_frag(At, ks, lane);
      v16h bk = load_frag(wkp, ks * 16 + nt, lane);
      v16h bv = load_frag(wvp, ks * 16 + nt, lane);
      ck = wmma16(a, bk, ck);
      cv = wmma16(a, bv, cv);
    }
    const int col = nt * 16 + (lane & 15);
    const int mh = (lane & 16) ? 8 : 0;
#pragma unroll
    for (int r = 0; r < 8; ++r) {
      const int n = n0 + mh + r;
      kf[((size_t)b * kN + n) * kD + col] = (_Float16)ck[r];
      const size_t vtile = ((size_t)b * (kN / 32) + (n >> 5)) * 16 + nt;
      vpack[(vtile << 9) + b_frag_off(n & 31, col & 15)] = (_Float16)cv[r];
    }
  }
}

// ---------------------------------------------------------------------------
// Kernel 2: LN(slots) fused with Q projection (* 1/sqrt(D)); q stored in
// A-fragment order for the logits GEMM.
// ---------------------------------------------------------------------------
__global__ __launch_bounds__(128) void slotln_q_kernel(
    const float* __restrict__ slots, const float* __restrict__ g,
    const float* __restrict__ be, const _Float16* __restrict__ wqp,
    _Float16* __restrict__ qpack) {
  __shared__ _Float16 At[8 * 512];
  __shared__ float red[16 * 8 * 2];
  const int b = blockIdx.x;
  layernorm16_to_lds_pack(slots + (size_t)b * kS * kD, g, be, At, red);
  const int wave = threadIdx.x >> 5, lane = threadIdx.x & 31;
#pragma unroll
  for (int t = 0; t < 4; ++t) {
    const int nt = wave * 4 + t;
    v8f c = {};
#pragma unroll
    for (int ks = 0; ks < 8; ++ks) {
      v16h a  = load_frag(At, ks, lane);
      v16h bw = load_frag(wqp, ks * 16 + nt, lane);
      c = wmma16(a, bw, c);
    }
    const int col = nt * 16 + (lane & 15);
    const int mh = (lane & 16) ? 8 : 0;
#pragma unroll
    for (int r = 0; r < 8; ++r) {
      const float qv = c[r] * 0.0625f;  // 1/sqrt(256)
      qpack[(((size_t)b * 8 + (col >> 5)) << 9) + a_frag_off(mh + r, col & 31)] =
          (_Float16)qv;
    }
  }
}

// ---------------------------------------------------------------------------
// Kernel 3a: logits = q @ K^T with fused softmax over the slot axis (S=16 ==
// tile M -> whole softmax axis lives inside one C column; combine the two lane
// halves with shfl_xor 16).  Stores attn (pre-renorm) in A-fragment order and
// accumulates per-slot row sums for the mean renormalization.
// ---------------------------------------------------------------------------
__global__ __launch_bounds__(128) void logits_softmax_kernel(
    const _Float16* __restrict__ qpack, const _Float16* __restrict__ kf,
    _Float16* __restrict__ apack, float* __restrict__ rowsum) {
  const int b = blockIdx.y;
  const int wave = threadIdx.x >> 5, lane = threadIdx.x & 31;
  const _Float16* qb = qpack + ((size_t)b * 8 << 9);
  const _Float16* kb = kf + (size_t)b * kN * kD;
#pragma unroll
  for (int t = 0; t < 4; ++t) {
    const int nt = blockIdx.x * 16 + wave * 4 + t;  // 256 n-tiles total
    const int n0 = nt * 16;
    v8f c = {};
#pragma unroll
    for (int ks = 0; ks < 8; ++ks) {
      v16h a  = load_frag(qb, ks, lane);
      v16h bt = load_bt_f16(kb, kD, n0, ks * 32, lane);
      c = wmma16(a, bt, c);
    }
    // softmax over M (slot) axis per column n
    float mx = c[0];
#pragma unroll
    for (int r = 1; r < 8; ++r) mx = fmaxf(mx, c[r]);
    mx = fmaxf(mx, __shfl_xor(mx, 16, 32));
    float e[8];
    float sum = 0.f;
#pragma unroll
    for (int r = 0; r < 8; ++r) { e[r] = __expf(c[r] - mx); sum += e[r]; }
    sum += __shfl_xor(sum, 16, 32);
    const float inv = 1.f / sum;
    const int col = n0 + (lane & 15);
    const int mh = (lane & 16) ? 8 : 0;
    const size_t atile = ((size_t)b * (kN / 32) + (col >> 5)) << 9;
#pragma unroll
    for (int r = 0; r < 8; ++r) {
      const float av = e[r] * inv;
      apack[atile + a_frag_off(mh + r, col & 31)] = (_Float16)av;
      float rs = av;
      rs += __shfl_xor(rs, 1, 32);
      rs += __shfl_xor(rs, 2, 32);
      rs += __shfl_xor(rs, 4, 32);
      rs += __shfl_xor(rs, 8, 32);
      if ((lane & 15) == 0) atomicAdd(&rowsum[b * kS + mh + r], rs);
    }
  }
}

// ---------------------------------------------------------------------------
// Kernel 3b: updates = renorm(attn) @ V  (split-K over N=4096, f32 atomics).
// Renorm scale applied as per-lane f16 splat on the loaded A fragment (lane ->
// slot row is invariant in the A layout).
// ---------------------------------------------------------------------------
__global__ __launch_bounds__(128) void updates_kernel(
    const _Float16* __restrict__ apack, const float* __restrict__ rowsum,
    const _Float16* __restrict__ vpack, float* __restrict__ updates) {
  const int b = blockIdx.y;
  const int kc = blockIdx.x;  // 8 chunks x 16 ksteps x 32 = 4096
  const int wave = threadIdx.x >> 5, lane = threadIdx.x & 31;
  const _Float16 hs =
      (_Float16)(1.f / (rowsum[b * kS + (lane & 15)] + 1e-8f));
  const _Float16* ab = apack + ((size_t)b * (kN / 32) << 9);
  const _Float16* vb = vpack + ((size_t)b * (kN / 32) * 16 << 9);
#pragma unroll
  for (int t = 0; t < 4; ++t) {
    const int nt = wave * 4 + t;
    v8f c = {};
    for (int ks = 0; ks < 16; ++ks) {
      const int kidx = kc * 16 + ks;
      __builtin_prefetch((const void*)(vb + ((size_t)((kidx + 1) * 16 + nt) << 9)),
                         0, 1);
      v16h a = load_frag(ab, kidx, lane);
#pragma unroll
      for (int i = 0; i < 16; ++i) a[i] = a[i] * hs;
      v16h bb = load_frag(vb, kidx * 16 + nt, lane);
      c = wmma16(a, bb, c);
    }
    const int col = nt * 16 + (lane & 15);
    const int mh = (lane & 16) ? 8 : 0;
#pragma unroll
    for (int r = 0; r < 8; ++r)
      atomicAdd(&updates[((size_t)b * kS + mh + r) * kD + col], c[r]);
  }
}

// ---------------------------------------------------------------------------
// Kernel 4: fused GRU — 6 WMMA GEMMs + elementwise gate math.
// ---------------------------------------------------------------------------
__global__ __launch_bounds__(128) void gru_kernel(
    const float* __restrict__ updates, const float* __restrict__ slots,
    const _Float16* __restrict__ wir, const _Float16* __restrict__ wiz,
    const _Float16* __restrict__ win, const _Float16* __restrict__ whr,
    const _Float16* __restrict__ whz, const _Float16* __restrict__ whn,
    const float* __restrict__ bir, const float* __restrict__ biz,
    const float* __restrict__ bin, const float* __restrict__ bhn,
    float* __restrict__ out) {
  const int b = blockIdx.x;
  const int wave = threadIdx.x >> 5, lane = threadIdx.x & 31;
  const float* ub = updates + (size_t)b * kS * kD;
  const float* sb = slots + (size_t)b * kS * kD;
#pragma unroll
  for (int t = 0; t < 4; ++t) {
    const int nt = wave * 4 + t;
    v8f cir = {}, ciz = {}, c_in = {}, c_hr = {}, c_hz = {}, c_hn = {};
#pragma unroll
    for (int ks = 0; ks < 8; ++ks) {
      v16h au = load_a_f32(ub + ks * 32, kD, lane);
      v16h as = load_a_f32(sb + ks * 32, kD, lane);
      const int wt = ks * 16 + nt;
      cir  = wmma16(au, load_frag(wir, wt, lane), cir);
      ciz  = wmma16(au, load_frag(wiz, wt, lane), ciz);
      c_in = wmma16(au, load_frag(win, wt, lane), c_in);
      c_hr = wmma16(as, load_frag(whr, wt, lane), c_hr);
      c_hz = wmma16(as, load_frag(whz, wt, lane), c_hz);
      c_hn = wmma16(as, load_frag(whn, wt, lane), c_hn);
    }
    const int col = nt * 16 + (lane & 15);
    const int mh = (lane & 16) ? 8 : 0;
#pragma unroll
    for (int r = 0; r < 8; ++r) {
      const size_t idx = ((size_t)b * kS + mh + r) * kD + col;
      const float rr = 1.f / (1.f + __expf(-(cir[r] + bir[col] + c_hr[r])));
      const float zz = 1.f / (1.f + __expf(-(ciz[r] + biz[col] + c_hz[r])));
      const float nn = tanhf(c_in[r] + bin[col] + rr * (c_hn[r] + bhn[col]));
      out[idx] = (1.f - zz) * nn + zz * slots[idx];
    }
  }
}

// ---------------------------------------------------------------------------
// Kernel 5: LN(gru_out) fused with MLP layer 1 + bias + ReLU; output stored in
// A-fragment order for the second MLP GEMM.
// ---------------------------------------------------------------------------
__global__ __launch_bounds__(128) void mlp1_kernel(
    const float* __restrict__ gruo, const float* __restrict__ g,
    const float* __restrict__ be, const _Float16* __restrict__ w1p,
    const float* __restrict__ b1, _Float16* __restrict__ hpack) {
  __shared__ _Float16 At[8 * 512];
  __shared__ float red[16 * 8 * 2];
  const int b = blockIdx.x;
  layernorm16_to_lds_pack(gruo + (size_t)b * kS * kD, g, be, At, red);
  const int wave = threadIdx.x >> 5, lane = threadIdx.x & 31;
  for (int t = 0; t < 16; ++t) {  // 64 n-tiles over 4 waves
    const int nt = wave * 16 + t;
    v8f c = {};
#pragma unroll
    for (int ks = 0; ks < 8; ++ks) {
      v16h a  = load_frag(At, ks, lane);
      v16h bw = load_frag(w1p, ks * 64 + nt, lane);
      c = wmma16(a, bw, c);
    }
    const int col = nt * 16 + (lane & 15);
    const int mh = (lane & 16) ? 8 : 0;
#pragma unroll
    for (int r = 0; r < 8; ++r) {
      const float v = c[r] + b1[col];
      hpack[(((size_t)b * (kM / 32) + (col >> 5)) << 9) +
            a_frag_off(mh + r, col & 31)] = (_Float16)(v > 0.f ? v : 0.f);
    }
  }
}

// ---------------------------------------------------------------------------
// Kernel 6: MLP layer 2 + bias + residual -> new slots (f32)
// ---------------------------------------------------------------------------
__global__ __launch_bounds__(128) void mlp2_kernel(
    const float* __restrict__ gruo, const _Float16* __restrict__ hpack,
    const _Float16* __restrict__ w2p, const float* __restrict__ b2,
    float* __restrict__ slots_out) {
  const int b = blockIdx.x;
  const int wave = threadIdx.x >> 5, lane = threadIdx.x & 31;
  const _Float16* hb = hpack + ((size_t)b * (kM / 32) << 9);
#pragma unroll
  for (int t = 0; t < 4; ++t) {
    const int nt = wave * 4 + t;
    v8f c = {};
    for (int ks = 0; ks < 32; ++ks) {  // K = 1024
      v16h a  = load_frag(hb, ks, lane);
      v16h bw = load_frag(w2p, ks * 16 + nt, lane);
      c = wmma16(a, bw, c);
    }
    const int col = nt * 16 + (lane & 15);
    const int mh = (lane & 16) ? 8 : 0;
#pragma unroll
    for (int r = 0; r < 8; ++r) {
      const size_t idx = ((size_t)b * kS + mh + r) * kD + col;
      slots_out[idx] = gruo[idx] + c[r] + b2[col];
    }
  }
}

// ---------------------------------------------------------------------------
// Host orchestration
// ---------------------------------------------------------------------------
extern "C" void kernel_launch(void* const* d_in, const int* in_sizes, int n_in,
                              void* d_out, int out_size, void* d_ws,
                              size_t ws_size, hipStream_t stream) {
  (void)in_sizes; (void)n_in; (void)ws_size;
  const float* slots_in = (const float*)d_in[0];
  const float* inputs   = (const float*)d_in[1];
  const float* ln_in_s  = (const float*)d_in[2];
  const float* ln_in_b  = (const float*)d_in[3];
  const float* w_q  = (const float*)d_in[4];
  const float* w_k  = (const float*)d_in[5];
  const float* w_v  = (const float*)d_in[6];
  const float* ln_q_s = (const float*)d_in[7];
  const float* ln_q_b = (const float*)d_in[8];
  const float* w_ir = (const float*)d_in[9];
  const float* b_ir = (const float*)d_in[10];
  const float* w_iz = (const float*)d_in[11];
  const float* b_iz = (const float*)d_in[12];
  const float* w_in = (const float*)d_in[13];
  const float* b_in = (const float*)d_in[14];
  const float* w_hr = (const float*)d_in[15];
  const float* w_hz = (const float*)d_in[16];
  const float* w_hn = (const float*)d_in[17];
  const float* b_hn = (const float*)d_in[18];
  const float* ln_m_s = (const float*)d_in[19];
  const float* ln_m_b = (const float*)d_in[20];
  const float* w_m1 = (const float*)d_in[21];
  const float* b_m1 = (const float*)d_in[22];
  const float* w_m2 = (const float*)d_in[23];
  const float* b_m2 = (const float*)d_in[24];

  char* ws = (char*)d_ws;
  size_t off = 0;
  auto alloc = [&](size_t bytes) -> char* {
    char* p = ws + off;
    off += (bytes + 255) & ~(size_t)255;
    return p;
  };
  _Float16* kf16      = (_Float16*)alloc((size_t)kB * kN * kD * 2);  // 64 MB
  _Float16* vpack     = (_Float16*)alloc((size_t)kB * kN * kD * 2);  // 64 MB
  float*    slots_cur = (float*)alloc((size_t)kB * kS * kD * 4);
  float*    gru_out   = (float*)alloc((size_t)kB * kS * kD * 4);
  float*    updates   = (float*)alloc((size_t)kB * kS * kD * 4);
  _Float16* qpack     = (_Float16*)alloc((size_t)kB * kS * kD * 2);
  _Float16* apack     = (_Float16*)alloc((size_t)kB * kS * kN * 2);  // 4 MB
  float*    rowsum    = (float*)alloc((size_t)kB * kS * 4);
  _Float16* hpack     = (_Float16*)alloc((size_t)kB * kS * kM * 2);
  _Float16* wqp  = (_Float16*)alloc((size_t)kD * kD * 2);
  _Float16* wkp  = (_Float16*)alloc((size_t)kD * kD * 2);
  _Float16* wvp  = (_Float16*)alloc((size_t)kD * kD * 2);
  _Float16* wirp = (_Float16*)alloc((size_t)kD * kD * 2);
  _Float16* wizp = (_Float16*)alloc((size_t)kD * kD * 2);
  _Float16* winp = (_Float16*)alloc((size_t)kD * kD * 2);
  _Float16* whrp = (_Float16*)alloc((size_t)kD * kD * 2);
  _Float16* whzp = (_Float16*)alloc((size_t)kD * kD * 2);
  _Float16* whnp = (_Float16*)alloc((size_t)kD * kD * 2);
  _Float16* wm1p = (_Float16*)alloc((size_t)kD * kM * 2);
  _Float16* wm2p = (_Float16*)alloc((size_t)kM * kD * 2);

  auto pack = [&](const float* s, _Float16* d, int K, int N) {
    pack_b_kernel<<<(K * N + 255) / 256, 256, 0, stream>>>(s, d, K, N);
  };
  pack(w_q, wqp, kD, kD);   pack(w_k, wkp, kD, kD);   pack(w_v, wvp, kD, kD);
  pack(w_ir, wirp, kD, kD); pack(w_iz, wizp, kD, kD); pack(w_in, winp, kD, kD);
  pack(w_hr, whrp, kD, kD); pack(w_hz, whzp, kD, kD); pack(w_hn, whnp, kD, kD);
  pack(w_m1, wm1p, kD, kM); pack(w_m2, wm2p, kM, kD);
  hipMemcpyAsync(slots_cur, slots_in, (size_t)kB * kS * kD * 4,
                 hipMemcpyDeviceToDevice, stream);

  // One-time: LN(inputs) + K/V projections
  ln_kv_kernel<<<kB * kN / 16, 128, 0, stream>>>(inputs, ln_in_s, ln_in_b, wkp,
                                                 wvp, kf16, vpack);

  for (int it = 0; it < 3; ++it) {
    slotln_q_kernel<<<kB, 128, 0, stream>>>(slots_cur, ln_q_s, ln_q_b, wqp,
                                            qpack);
    hipMemsetAsync(rowsum, 0, (size_t)kB * kS * 4, stream);
    logits_softmax_kernel<<<dim3(16, kB), 128, 0, stream>>>(qpack, kf16, apack,
                                                            rowsum);
    hipMemsetAsync(updates, 0, (size_t)kB * kS * kD * 4, stream);
    updates_kernel<<<dim3(8, kB), 128, 0, stream>>>(apack, rowsum, vpack,
                                                    updates);
    gru_kernel<<<kB, 128, 0, stream>>>(updates, slots_cur, wirp, wizp, winp,
                                       whrp, whzp, whnp, b_ir, b_iz, b_in, b_hn,
                                       gru_out);
    mlp1_kernel<<<kB, 128, 0, stream>>>(gru_out, ln_m_s, ln_m_b, wm1p, b_m1,
                                        hpack);
    mlp2_kernel<<<kB, 128, 0, stream>>>(gru_out, hpack, wm2p, b_m2, slots_cur);
  }

  hipMemcpyAsync(d_out, slots_cur, (size_t)out_size * 4,
                 hipMemcpyDeviceToDevice, stream);
}
